// _CausalAttnLayer_54855322304759
// MI455X (gfx1250) — compile-verified, matrix operand order checked by reading, not run on hardware
//
#include <hip/hip_runtime.h>
#include <hip/hip_bf16.h>
#include <math.h>
#include <stdint.h>

// ---------------- model constants (match reference) ----------------
#define DMODEL 1024
#define NH     16
#define HDIM   64
#define FFDIM  4096
#define BATCH  2
#define SEQ    2048
#define MROWS  (BATCH * SEQ)   // 4096 token rows

typedef __attribute__((ext_vector_type(16))) __bf16 v16bf;
typedef __attribute__((ext_vector_type(8)))  float  v8f;
typedef __attribute__((ext_vector_type(4)))  unsigned int u32x4;
typedef __attribute__((ext_vector_type(8)))  unsigned int u32x8;

union FragU { v16bf v; int4 i[2]; };

__device__ __forceinline__ v8f wmma_bf16(v16bf a, v16bf b, v8f c) {
  // D = A(16x32 bf16) x B(32x16 bf16) + C(16x16 f32)
  return __builtin_amdgcn_wmma_f32_16x16x32_bf16(false, a, false, b,
                                                 (short)0, c, false, false);
}

// ---- CDNA5 async global->LDS copy (ASYNCcnt) ----
__device__ __forceinline__ void async_ld_b128(unsigned lds_off, const void* gaddr) {
  asm volatile("global_load_async_to_lds_b128 %0, %1, off"
               :: "v"(lds_off), "v"((unsigned long long)(uintptr_t)gaddr)
               : "memory");
}
__device__ __forceinline__ void wait_asynccnt0() {
  asm volatile("s_wait_asynccnt 0x0" ::: "memory");
}
__device__ __forceinline__ void wait_tensorcnt0() {
  asm volatile("s_wait_tensorcnt 0x0" ::: "memory");
}

// ---- CDNA5 Tensor Data Mover: 2D tile load (global -> LDS), wave-level ----
// D# group0: count=1, lds_addr, 57-bit global addr, type=2 ("image")
// D# group1: data_size=2B, tensor_dim0/1, tile_dim0/1, tensor_dim0_stride
__device__ __forceinline__ void tdm_load_2d(unsigned lds_addr, const void* gaddr,
                                            unsigned tile_w, unsigned tile_h,
                                            unsigned stride_elems) {
  const unsigned long long ga = (unsigned long long)(uintptr_t)gaddr;
  u32x4 g0;
  g0.x = 1u;                                  // count=1 (valid user descriptor)
  g0.y = lds_addr;                            // LDS byte address
  g0.z = (unsigned)ga;                        // global_addr[31:0]
  g0.w = (unsigned)(ga >> 32) | (2u << 30);   // global_addr[56:32] | type=2
  u32x8 g1;
  g1.s0 = (1u << 16);                         // data_size=1 (2 bytes), mask=0
  g1.s1 = tile_w << 16;                       // tensor_dim0[15:0]  (bits 63:48)
  g1.s2 = tile_h << 16;                       // tensor_dim1[15:0]  (bits 95:80)
  g1.s3 = tile_w << 16;                       // tile_dim0          (bits 127:112)
  g1.s4 = tile_h;                             // tile_dim1          (bits 143:128)
  g1.s5 = stride_elems;                       // tensor_dim0_stride (bits 191:160)
  g1.s6 = 0u;
  g1.s7 = 0u;
  asm volatile("tensor_load_to_lds %0, %1" :: "s"(g0), "s"(g1) : "memory");
}

// ---------------- fp32 -> bf16 cast (weights) ----------------
__global__ __launch_bounds__(256)
void cast_f32_to_bf16(const float* __restrict__ in, __bf16* __restrict__ out, int n4) {
  int i = blockIdx.x * 256 + threadIdx.x;
  if (i < n4) {
    float4 v = ((const float4*)in)[i];
    __bf16* o = out + (size_t)i * 4;
    o[0] = (__bf16)v.x; o[1] = (__bf16)v.y; o[2] = (__bf16)v.z; o[3] = (__bf16)v.w;
  }
}

// ---------------- LayerNorm (fp32 in, bf16 out), one block per row ----------------
__global__ __launch_bounds__(256)
void ln_bf16_kernel(const float* __restrict__ x, const float* __restrict__ g,
                    const float* __restrict__ b, __bf16* __restrict__ out) {
  __shared__ float rs[256], rs2[256];
  const int row = blockIdx.x, tid = threadIdx.x;
  const float4 v = ((const float4*)(x + (size_t)row * DMODEL))[tid];
  rs[tid]  = v.x + v.y + v.z + v.w;
  rs2[tid] = v.x * v.x + v.y * v.y + v.z * v.z + v.w * v.w;
  __syncthreads();
  for (int off = 128; off > 0; off >>= 1) {
    if (tid < off) { rs[tid] += rs[tid + off]; rs2[tid] += rs2[tid + off]; }
    __syncthreads();
  }
  const float mean = rs[0] * (1.0f / DMODEL);
  const float var  = rs2[0] * (1.0f / DMODEL) - mean * mean;
  const float inv  = rsqrtf(var + 1e-5f);
  const float4 gg = ((const float4*)g)[tid];
  const float4 bb = ((const float4*)b)[tid];
  __bf16* o = out + (size_t)row * DMODEL + tid * 4;
  o[0] = (__bf16)((v.x - mean) * inv * gg.x + bb.x);
  o[1] = (__bf16)((v.y - mean) * inv * gg.y + bb.y);
  o[2] = (__bf16)((v.z - mean) * inv * gg.z + bb.z);
  o[3] = (__bf16)((v.w - mean) * inv * gg.w + bb.w);
}

// ---------------- bf16 GEMM: C = A[MxK] * B[KxN] (+bias, gelu, residual) ----------------
// Block tile 128x64, 256 threads = 8 waves; each wave owns a 32x32 sub-tile.
// A tile staged via CDNA5 async global->LDS; B tile transposed through VGPRs.
__global__ __launch_bounds__(256)
void gemm_bf16_kernel(const __bf16* __restrict__ A, const __bf16* __restrict__ Bw,
                      const float* __restrict__ bias, const float* __restrict__ residual,
                      float* __restrict__ out_f, __bf16* __restrict__ out_b,
                      int M, int N, int K, int do_gelu) {
  __shared__ __attribute__((aligned(16))) __bf16 As[128 * 32];  // [m][k]
  __shared__ __attribute__((aligned(16))) __bf16 Bs[64 * 32];   // [n][k] (transposed)

  const int tid = threadIdx.x;
  const int wv = tid >> 5, lane = tid & 31;
  const int hf = lane >> 4, l16 = lane & 15;
  const int wr = wv >> 1, wc = wv & 1;           // 4x2 wave grid
  const int m0 = blockIdx.y * 128, n0 = blockIdx.x * 64;

  v8f acc[2][2] = {};

  const int ar = tid >> 1, ah = tid & 1;         // A staging: row, 16-elem half
  const int bk = tid >> 3, bg = tid & 7;         // B staging: k row, 8-col group

  for (int k0 = 0; k0 < K; k0 += 32) {
    // stage A tile 128x32: async DMA straight into LDS (no VGPR round-trip)
    {
      const __bf16* asrc = A + (size_t)(m0 + ar) * K + k0 + ah * 16;
      async_ld_b128((unsigned)(uintptr_t)&As[ar * 32 + ah * 16],     asrc);
      async_ld_b128((unsigned)(uintptr_t)&As[ar * 32 + ah * 16 + 8], asrc + 8);
    }
    // stage B tile 32x64 transposed into Bs[n][k]
    {
      int4 brow = *(const int4*)(Bw + (size_t)(k0 + bk) * N + n0 + bg * 8);
      const __bf16* bpv = (const __bf16*)&brow;
      #pragma unroll
      for (int j = 0; j < 8; ++j) Bs[(bg * 8 + j) * 32 + bk] = bpv[j];
    }
    wait_asynccnt0();
    __syncthreads();

    if (k0 + 32 < K)
      __builtin_prefetch(A + (size_t)(m0 + ar) * K + k0 + 32, 0, 0);

    FragU af[2], bf2[2];
    #pragma unroll
    for (int mi = 0; mi < 2; ++mi) {
      const __bf16* p = &As[(wr * 32 + mi * 16 + l16) * 32];
      af[mi].i[0] = *(const int4*)(p + hf * 8);
      af[mi].i[1] = *(const int4*)(p + 16 + hf * 8);
    }
    #pragma unroll
    for (int ni = 0; ni < 2; ++ni) {
      const __bf16* p = &Bs[(wc * 32 + ni * 16 + l16) * 32];
      bf2[ni].i[0] = *(const int4*)(p + hf * 8);
      bf2[ni].i[1] = *(const int4*)(p + 16 + hf * 8);
    }
    #pragma unroll
    for (int mi = 0; mi < 2; ++mi)
      #pragma unroll
      for (int ni = 0; ni < 2; ++ni)
        acc[mi][ni] = wmma_bf16(af[mi].v, bf2[ni].v, acc[mi][ni]);

    __syncthreads();
  }

  // epilogue: bias (+gelu) (+residual), store fp32 and/or bf16
  #pragma unroll
  for (int mi = 0; mi < 2; ++mi)
    #pragma unroll
    for (int ni = 0; ni < 2; ++ni) {
      const int n = n0 + wc * 32 + ni * 16 + l16;
      const float bsv = bias[n];
      #pragma unroll
      for (int r = 0; r < 8; ++r) {
        const int m = m0 + wr * 32 + mi * 16 + r + hf * 8;
        float v = acc[mi][ni][r] + bsv;
        if (do_gelu) v = 0.5f * v * (1.0f + erff(v * 0.70710678118654752f));
        if (residual) v += residual[(size_t)m * N + n];
        if (out_f) out_f[(size_t)m * N + n] = v;
        if (out_b) out_b[(size_t)m * N + n] = (__bf16)v;
      }
    }
}

// ---------------- causal flash attention ----------------
// grid (SEQ/64, NH, BATCH), 128 threads (4 waves); wave w owns q rows q0+16w..+15.
// K/V tiles (64x64 bf16) fetched by the Tensor Data Mover, issued by wave 0.
__global__ __launch_bounds__(128)
void attn_kernel(const __bf16* __restrict__ qkv, __bf16* __restrict__ attn_out) {
  __shared__ __attribute__((aligned(16))) __bf16 Kt[64 * 64];      // [kv][d]
  __shared__ __attribute__((aligned(16))) __bf16 Vt[64 * 64];      // [kv][d]
  __shared__ __attribute__((aligned(16))) __bf16 Pb[4][16 * 64];   // per-wave P

  const int tid = threadIdx.x;
  const int wv = tid >> 5, lane = tid & 31;
  const int hf = lane >> 4, l16 = lane & 15;
  const int bidx = blockIdx.z, hh = blockIdx.y;
  const int q0 = blockIdx.x * 64;
  const size_t rowstride = 3 * DMODEL;

  // Q fragments (persistent): rows q0+16w+l16, dims 0..63 (two K=32 fragments)
  FragU qf[2];
  {
    const int qrow = q0 + wv * 16 + l16;
    const __bf16* qp = qkv + (size_t)(bidx * SEQ + qrow) * rowstride + hh * HDIM;
    #pragma unroll
    for (int kk = 0; kk < 2; ++kk) {
      qf[kk].i[0] = *(const int4*)(qp + kk * 32 + hf * 8);
      qf[kk].i[1] = *(const int4*)(qp + kk * 32 + 16 + hf * 8);
    }
  }

  v8f o[4] = {};
  float mrun[8], lrun[8];
  #pragma unroll
  for (int r = 0; r < 8; ++r) { mrun[r] = -1e30f; lrun[r] = 0.0f; }

  const int nkt = (q0 >> 6) + 1;           // causal: only kv tiles <= q tile

  for (int kt = 0; kt < nkt; ++kt) {
    const int k0 = kt * 64;
    // TDM: one wave issues two 2D tile DMAs (K and V), 64 rows x 128 B, stride 6 KB
    if (wv == 0) {
      const __bf16* kp = qkv + (size_t)(bidx * SEQ + k0) * rowstride + DMODEL + hh * HDIM;
      tdm_load_2d((unsigned)(uintptr_t)&Kt[0], kp,          64, 64, 3 * DMODEL);
      tdm_load_2d((unsigned)(uintptr_t)&Vt[0], kp + DMODEL, 64, 64, 3 * DMODEL);
      wait_tensorcnt0();
    }
    __syncthreads();

    // S = Q K^T  (B fragment cols = kv index -> rows of Kt; contiguous in d)
    v8f s[4];
    #pragma unroll
    for (int ni = 0; ni < 4; ++ni) {
      v8f a = {};
      const __bf16* kc = &Kt[(ni * 16 + l16) * 64];
      #pragma unroll
      for (int kk = 0; kk < 2; ++kk) {
        FragU bfr;
        bfr.i[0] = *(const int4*)(kc + kk * 32 + hf * 8);
        bfr.i[1] = *(const int4*)(kc + kk * 32 + 16 + hf * 8);
        a = wmma_bf16(qf[kk].v, bfr.v, a);
      }
      s[ni] = a;
    }

    // online softmax (per lane: rows r + hf*8, col l16; reduce across 16-lane group)
    #pragma unroll
    for (int r = 0; r < 8; ++r) {
      const int q = q0 + wv * 16 + r + hf * 8;
      float rowmax = -1e30f;
      #pragma unroll
      for (int ni = 0; ni < 4; ++ni) {
        const int kc = k0 + ni * 16 + l16;
        float v = s[ni][r] * 0.125f;             // 1/sqrt(64)
        if (kc > q) v = -1e30f;                  // causal mask
        s[ni][r] = v;
        rowmax = fmaxf(rowmax, v);
      }
      #pragma unroll
      for (int off = 1; off < 16; off <<= 1)
        rowmax = fmaxf(rowmax, __shfl_xor(rowmax, off, 32));
      const float mnew  = fmaxf(mrun[r], rowmax);
      const float alpha = __expf(mrun[r] - mnew);
      float psum = 0.0f;
      #pragma unroll
      for (int ni = 0; ni < 4; ++ni) {
        float p = __expf(s[ni][r] - mnew);
        s[ni][r] = p;
        psum += p;
      }
      #pragma unroll
      for (int off = 1; off < 16; off <<= 1)
        psum += __shfl_xor(psum, off, 32);
      lrun[r] = lrun[r] * alpha + psum;
      mrun[r] = mnew;
      #pragma unroll
      for (int nb = 0; nb < 4; ++nb) o[nb][r] *= alpha;
    }

    // write P to wave-private LDS (transposes C-layout -> A-fragment layout)
    #pragma unroll
    for (int ni = 0; ni < 4; ++ni)
      #pragma unroll
      for (int r = 0; r < 8; ++r)
        Pb[wv][(r + hf * 8) * 64 + ni * 16 + l16] = (__bf16)s[ni][r];

    // O += P V
    #pragma unroll
    for (int kk = 0; kk < 2; ++kk) {
      FragU pa;
      const __bf16* pr = &Pb[wv][l16 * 64 + kk * 32];
      pa.i[0] = *(const int4*)(pr + hf * 8);
      pa.i[1] = *(const int4*)(pr + 16 + hf * 8);
      #pragma unroll
      for (int nb = 0; nb < 4; ++nb) {
        v16bf vb;
        const __bf16* vbase = &Vt[(size_t)kk * 32 * 64 + nb * 16 + l16];
        #pragma unroll
        for (int e = 0; e < 16; ++e) {
          const int kv = ((e >> 3) << 4) + (hf << 3) + (e & 7);  // 16-bit frag k-map
          vb[e] = vbase[kv * 64];
        }
        o[nb] = wmma_bf16(pa.v, vb, o[nb]);
      }
    }
    __syncthreads();
  }

  // normalize and store bf16 (b, q, h, d) -> (b*t, D)
  #pragma unroll
  for (int nb = 0; nb < 4; ++nb)
    #pragma unroll
    for (int r = 0; r < 8; ++r) {
      const int q = q0 + wv * 16 + r + hf * 8;
      const float v = o[nb][r] / lrun[r];
      attn_out[(size_t)(bidx * SEQ + q) * DMODEL + hh * HDIM + nb * 16 + l16] = (__bf16)v;
    }
}

// ---------------- host orchestration ----------------
extern "C" void kernel_launch(void* const* d_in, const int* in_sizes, int n_in,
                              void* d_out, int out_size, void* d_ws, size_t ws_size,
                              hipStream_t stream) {
  (void)in_sizes; (void)n_in; (void)out_size; (void)ws_size;
  const float* x    = (const float*)d_in[0];
  const float* g1   = (const float*)d_in[1];
  const float* b1   = (const float*)d_in[2];
  const float* Wqkv = (const float*)d_in[3];
  const float* bqkv = (const float*)d_in[4];
  const float* Wo   = (const float*)d_in[5];
  const float* bo   = (const float*)d_in[6];
  const float* g2   = (const float*)d_in[7];
  const float* b2   = (const float*)d_in[8];
  const float* W1   = (const float*)d_in[9];
  const float* b1f  = (const float*)d_in[10];
  const float* W2   = (const float*)d_in[11];
  const float* b2f  = (const float*)d_in[12];
  float* out = (float*)d_out;

  char* ws = (char*)d_ws;
  size_t off = 0;
  auto alloc = [&](size_t bytes) -> char* {
    char* p = ws + off;
    off += (bytes + 255) & ~(size_t)255;
    return p;
  };
  __bf16* wqkv_b = (__bf16*)alloc((size_t)DMODEL * 3 * DMODEL * 2);
  __bf16* wo_b   = (__bf16*)alloc((size_t)DMODEL * DMODEL * 2);
  __bf16* w1_b   = (__bf16*)alloc((size_t)DMODEL * FFDIM * 2);
  __bf16* w2_b   = (__bf16*)alloc((size_t)FFDIM * DMODEL * 2);
  __bf16* h1_b   = (__bf16*)alloc((size_t)MROWS * DMODEL * 2);
  __bf16* qkv_b  = (__bf16*)alloc((size_t)MROWS * 3 * DMODEL * 2);
  __bf16* attn_b = (__bf16*)alloc((size_t)MROWS * DMODEL * 2);
  float*  x2     = (float*)alloc((size_t)MROWS * DMODEL * 4);
  __bf16* h2_b   = (__bf16*)alloc((size_t)MROWS * DMODEL * 2);
  __bf16* ff_b   = (__bf16*)alloc((size_t)MROWS * FFDIM * 2);

  // 1. weights -> bf16
  const int nqkv = DMODEL * 3 * DMODEL / 4, nwo = DMODEL * DMODEL / 4,
            nw1 = DMODEL * FFDIM / 4, nw2 = FFDIM * DMODEL / 4;
  cast_f32_to_bf16<<<(nqkv + 255) / 256, 256, 0, stream>>>(Wqkv, wqkv_b, nqkv);
  cast_f32_to_bf16<<<(nwo  + 255) / 256, 256, 0, stream>>>(Wo,   wo_b,   nwo);
  cast_f32_to_bf16<<<(nw1  + 255) / 256, 256, 0, stream>>>(W1,   w1_b,   nw1);
  cast_f32_to_bf16<<<(nw2  + 255) / 256, 256, 0, stream>>>(W2,   w2_b,   nw2);

  // 2. LN1 -> bf16
  ln_bf16_kernel<<<MROWS, 256, 0, stream>>>(x, g1, b1, h1_b);

  // 3. qkv = h1 @ Wqkv + bqkv   (bf16 out)
  gemm_bf16_kernel<<<dim3(3 * DMODEL / 64, MROWS / 128), 256, 0, stream>>>(
      h1_b, wqkv_b, bqkv, nullptr, nullptr, qkv_b, MROWS, 3 * DMODEL, DMODEL, 0);

  // 4. causal flash attention
  attn_kernel<<<dim3(SEQ / 64, NH, BATCH), 128, 0, stream>>>(qkv_b, attn_b);

  // 5. x2 = x + attn @ Wo + bo   (fp32 out)
  gemm_bf16_kernel<<<dim3(DMODEL / 64, MROWS / 128), 256, 0, stream>>>(
      attn_b, wo_b, bo, x, x2, nullptr, MROWS, DMODEL, DMODEL, 0);

  // 6. LN2 -> bf16
  ln_bf16_kernel<<<MROWS, 256, 0, stream>>>(x2, g2, b2, h2_b);

  // 7. ff = gelu(h2 @ W1 + b1f)   (bf16 out)
  gemm_bf16_kernel<<<dim3(FFDIM / 64, MROWS / 128), 256, 0, stream>>>(
      h2_b, w1_b, b1f, nullptr, nullptr, ff_b, MROWS, FFDIM, DMODEL, 1);

  // 8. out = x2 + ff @ W2 + b2f   (fp32 out)
  gemm_bf16_kernel<<<dim3(DMODEL / 64, MROWS / 128), 256, 0, stream>>>(
      ff_b, w2_b, b2f, x2, out, nullptr, MROWS, DMODEL, FFDIM, 0);
}
